// VectorQuantizer_79706003079760
// MI455X (gfx1250) — matrix operand order repaired and checked by reading, not compiled, not run on hardware
//
#include <hip/hip_runtime.h>

typedef __attribute__((ext_vector_type(16))) _Float16 v16h;
typedef __attribute__((ext_vector_type(8)))  _Float16 v8h;
typedef __attribute__((ext_vector_type(8)))  float    v8f;

#define K_EMB   1024
#define D_EMB   64
#define N_ROWS  131072          // 32*64*64 spatial vectors
#define HW      4096            // 64*64
#define OUT_ELEMS 8388608       // N_ROWS * D_EMB

// ---------------------------------------------------------------------------
// Prep 1: repack embedding [1024,64] f32 -> f16 WMMA B-fragments.
// Layout (half units): idx = t*1024 + f*512 + g*256 + lane*8 + w
//   t = k-tile (0..63), f = which 32-wide K chunk (d/32), g = 8-half chunk,
//   lane = wave lane, w = half within chunk.
// B-fragment mapping (ISA 7.12.2, 16-bit B 32x16): column n = lane&15,
//   K-index (= d here) = f*32 + g*16 + (lane>>4)*8 + w.
// ---------------------------------------------------------------------------
__global__ void vq_prep_frag(const float* __restrict__ emb,
                             _Float16* __restrict__ efrag) {
    int i    = blockIdx.x * blockDim.x + threadIdx.x;   // 0..65535
    int w    = i & 7;
    int lane = (i >> 3) & 31;
    int g    = (i >> 8) & 1;
    int f    = (i >> 9) & 1;
    int t    = i >> 10;
    int n    = lane & 15;
    int hi   = lane >> 4;
    int d    = f * 32 + g * 16 + hi * 8 + w;
    int kk   = t * 16 + n;
    efrag[i] = (_Float16)emb[kk * D_EMB + d];
}

// Prep 2: ||e_k||^2 per codebook row.
__global__ void vq_prep_norm(const float* __restrict__ emb,
                             float* __restrict__ norms) {
    int k = blockIdx.x * blockDim.x + threadIdx.x;
    if (k < K_EMB) {
        float s = 0.f;
        #pragma unroll
        for (int d = 0; d < D_EMB; ++d) {
            float e = emb[k * D_EMB + d];
            s += e * e;
        }
        norms[k] = s;
    }
}

// ---------------------------------------------------------------------------
// Main: per wave, one 16-row tile of x. Score tile via WMMA against all 64
// k-tiles, argmax(x.e - 0.5||e||^2) with index tie-break, gather + write
// straight-through output, accumulate loss partial per block.
// ---------------------------------------------------------------------------
__global__ __launch_bounds__(256)
void vq_main(const float* __restrict__ lat,       // [32,64,64,64]
             const float* __restrict__ emb,       // [1024,64]
             const _Float16* __restrict__ efrag,  // repacked f16 fragments
             const float* __restrict__ norms,     // [1024]
             float* __restrict__ out,             // [32,64,64,64]
             float* __restrict__ partials) {      // [gridDim.x]
    __shared__ int   s_idx[8 * 16];
    __shared__ float s_lsum[8];

    const int lane = threadIdx.x & 31;
    const int wave = threadIdx.x >> 5;
    const int m    = lane & 15;     // row within tile (A) / column (B,C)
    const int hi   = lane >> 4;

    const int tile = blockIdx.x * 8 + wave;       // 0..8191
    const int n0   = tile * 16;
    const int b    = n0 >> 12;                    // batch
    const int hw0  = n0 & (HW - 1);               // spatial offset

    const float* xbase = lat + (size_t)b * (D_EMB * HW) + hw0 + m;

    // ---- Load A-fragments (x rows) once; keep f32 copies for loss/output.
    // A mapping (ISA 7.12.2, 16-bit A 16x32): lane row M = lane&15,
    //   half h -> K = hi*8 + (h<8 ? h : h+8), fragment f adds +32.
    float x32[32];
    v16h a0, a1;
    #pragma unroll
    for (int h = 0; h < 16; ++h) {
        int d0 = hi * 8 + ((h < 8) ? h : h + 8);
        float v0 = xbase[(size_t)d0 * HW];
        float v1 = xbase[(size_t)(d0 + 32) * HW];
        x32[h]      = v0;
        x32[16 + h] = v1;
        a0[h] = (_Float16)v0;
        a1[h] = (_Float16)v1;
    }

    // ---- Scan all 64 k-tiles. C/D layout: lane holds col m, rows j + 8*hi.
    float best[8];
    int   bidx[8];
    #pragma unroll
    for (int j = 0; j < 8; ++j) { best[j] = -3.4e38f; bidx[j] = 0; }

    const v8h* fb = (const v8h*)efrag + lane;     // + t*128 (+32/+64/+96)

    #pragma unroll 2
    for (int t = 0; t < 64; ++t) {
        const v8h* p  = fb + t * 128;
        v8h b00 = p[0];      // f=0,g=0
        v8h b01 = p[32];     // f=0,g=1
        v8h b10 = p[64];     // f=1,g=0
        v8h b11 = p[96];     // f=1,g=1
        v16h fb0 = __builtin_shufflevector(b00, b01,
                     0,1,2,3,4,5,6,7,8,9,10,11,12,13,14,15);
        v16h fb1 = __builtin_shufflevector(b10, b11,
                     0,1,2,3,4,5,6,7,8,9,10,11,12,13,14,15);

        float cinit = -0.5f * norms[t * 16 + m];
        v8f acc;
        #pragma unroll
        for (int j = 0; j < 8; ++j) acc[j] = cinit;

        acc = __builtin_amdgcn_wmma_f32_16x16x32_f16(
                  false, a0, false, fb0, (short)0, acc, false, false);
        acc = __builtin_amdgcn_wmma_f32_16x16x32_f16(
                  false, a1, false, fb1, (short)0, acc, false, false);

        int k = t * 16 + m;
        #pragma unroll
        for (int j = 0; j < 8; ++j) {
            float v = acc[j];
            if (v > best[j]) { best[j] = v; bidx[j] = k; }   // strict >: first k wins
        }
    }

    // ---- Reduce across the 16 lanes holding the same rows (butterfly).
    #pragma unroll
    for (int j = 0; j < 8; ++j) {
        float v = best[j];
        int   i = bidx[j];
        #pragma unroll
        for (int mask = 1; mask < 16; mask <<= 1) {
            float ov = __shfl_xor(v, mask, 32);
            int   oi = __shfl_xor(i, mask, 32);
            if (ov > v || (ov == v && oi < i)) { v = ov; i = oi; }
        }
        bidx[j] = i;
    }
    if (m == 0) {
        #pragma unroll
        for (int j = 0; j < 8; ++j) s_idx[wave * 16 + hi * 8 + j] = bidx[j];
    }
    __syncthreads();
    int myidx = s_idx[wave * 16 + m];

    // ---- Gather codebook row (f32), write straight-through out, loss.
    const float* erow  = emb + (size_t)myidx * D_EMB;
    float*       obase = out + (size_t)b * (D_EMB * HW) + hw0 + m;
    float lsum = 0.f;
    #pragma unroll
    for (int h = 0; h < 32; ++h) {
        int f  = h >> 4;
        int hh = h & 15;
        int d  = f * 32 + hi * 8 + ((hh < 8) ? hh : hh + 8);
        float q    = erow[d];
        float x    = x32[h];
        float diff = q - x;
        lsum += diff * diff;
        obase[(size_t)d * HW] = x + diff;   // x + stop_grad(q - x)
    }

    // ---- Deterministic per-block loss partial.
    #pragma unroll
    for (int mask = 1; mask < 32; mask <<= 1)
        lsum += __shfl_xor(lsum, mask, 32);
    if (lane == 0) s_lsum[wave] = lsum;
    __syncthreads();
    if (threadIdx.x == 0) {
        float s = 0.f;
        #pragma unroll
        for (int w2 = 0; w2 < 8; ++w2) s += s_lsum[w2];
        partials[blockIdx.x] = s;
    }
}

// ---------------------------------------------------------------------------
// Finalize: reduce 1024 partials, scale by (1+BETA)/numel, write loss scalar.
// ---------------------------------------------------------------------------
__global__ void vq_finalize(const float* __restrict__ partials,
                            float* __restrict__ loss) {
    __shared__ float sm[256];
    float s = 0.f;
    for (int i = threadIdx.x; i < 1024; i += 256) s += partials[i];
    sm[threadIdx.x] = s;
    __syncthreads();
    for (int off = 128; off > 0; off >>= 1) {
        if (threadIdx.x < off) sm[threadIdx.x] += sm[threadIdx.x + off];
        __syncthreads();
    }
    if (threadIdx.x == 0) loss[0] = sm[0] * (1.25f / (float)OUT_ELEMS);
}

extern "C" void kernel_launch(void* const* d_in, const int* in_sizes, int n_in,
                              void* d_out, int out_size, void* d_ws, size_t ws_size,
                              hipStream_t stream) {
    const float* lat = (const float*)d_in[0];   // latents [32,64,64,64] f32
    const float* emb = (const float*)d_in[1];   // embedding [1024,64] f32
    float* out = (float*)d_out;                 // [32,64,64,64] f32 + loss

    // Workspace layout
    _Float16* efrag    = (_Float16*)d_ws;                       // 131072 B
    float*    norms    = (float*)((char*)d_ws + 131072);        //   4096 B
    float*    partials = (float*)((char*)d_ws + 135168);        //   4096 B

    vq_prep_frag<<<256, 256, 0, stream>>>(emb, efrag);
    vq_prep_norm<<<4,   256, 0, stream>>>(emb, norms);
    vq_main<<<1024, 256, 0, stream>>>(lat, emb, efrag, norms, out, partials);
    vq_finalize<<<1, 256, 0, stream>>>(partials, out + OUT_ELEMS);
}
